// GaussianBlurLayer1_14053132992835
// MI455X (gfx1250) — compile-verified, matrix operand order checked by reading, not run on hardware
//
#include <hip/hip_runtime.h>

// Problem constants (from reference)
#define R_  60
#define K_  121
#define B_  64
#define C_  3
#define H_  256
#define W_  256
#define HP_ 376             // padded extent (H + 2R)
#define IP_ 384             // i-dimension of intermediate, tile-aligned (24*16)
#define LSH 408             // hblur LDS row stride in f16 (>=400 needed; banks: 12r mod 64)
#define LSV 424             // vblur LDS row stride in f16 (>=400 needed; banks: 20r mod 64)
#define TFS 512             // tap-fragment table: 32 lanes * 16 elems per (b,kk)

typedef _Float16 half_t;
typedef __attribute__((ext_vector_type(16))) _Float16 v16h;
typedef __attribute__((ext_vector_type(8)))  _Float16 v8h;
typedef __attribute__((ext_vector_type(8)))  float    v8f;

// numpy/jax 'reflect' (mirror, edge not repeated)
__device__ __forceinline__ int reflect_idx(int i, int n) {
  i = (i < 0) ? -i : i;
  i = (i >= n) ? (2 * n - 2 - i) : i;
  return i;
}

// 16-bit WMMA A/B fragment K-index for element e (0..15), lane (0..31):
// VGPRs 0-3 hold K = hi*8 + 0..7, VGPRs 4-7 hold K = 16 + hi*8 + 0..7
__device__ __forceinline__ int frag_k(int lane, int e) {
  return ((e >> 3) << 4) + ((lane >> 4) << 3) + (e & 7);
}

// ---------------------------------------------------------------------------
// Kernel 1: normalized taps, pre-packed into WMMA fragment layout:
// tf[b][kk][lane][e] = k[32*kk + frag_k(lane,e) - (lane&15)]  (0 outside [0,121))
// The same table is the B operand of hblur and the B operand of vblur.
// ---------------------------------------------------------------------------
__global__ __launch_bounds__(256) void build_taps(const int* __restrict__ fwd_steps,
                                                  const float* __restrict__ blur_sigmas,
                                                  half_t* __restrict__ tf) {
  const int b = blockIdx.x;
  const int t = threadIdx.x;
  __shared__ float sw[K_];
  __shared__ float ssum;

  const float sigma = blur_sigmas[fwd_steps[b]];
  const float r = ceilf(3.0f * sigma);
  if (t < K_) {
    float o = (float)(t - R_);
    float w = __expf(-0.5f * (o / sigma) * (o / sigma));
    sw[t] = (fabsf(o) <= r) ? w : 0.0f;
  }
  __syncthreads();
  if (t == 0) {
    float s = 0.0f;
    for (int i = 0; i < K_; ++i) s += sw[i];
    ssum = s;
  }
  __syncthreads();
  const float inv = 1.0f / ssum;
  for (int idx = t; idx < 5 * TFS; idx += 256) {
    int kk = idx / TFS, rem = idx - kk * TFS;
    int lane = rem >> 4, e = rem & 15;
    int tapix = 32 * kk + frag_k(lane, e) - (lane & 15);   // in [-15, 159]
    float v = (tapix >= 0 && tapix < K_) ? sw[tapix] * inv : 0.0f;
    tf[(size_t)(b * 5 + kk) * TFS + rem] = (half_t)v;
  }
}

// ---------------------------------------------------------------------------
// Kernel 2: horizontal pass. Y1t[w][i] = sum_j Xp[i, j] * k[j - w]  (transposed!)
// Stage 16 reflect-padded rows to LDS once; A fragments = 2x ds_load_b128/lane;
// B = preloaded tap fragments. D rows are contiguous in i -> one v8h store.
// ---------------------------------------------------------------------------
__global__ __launch_bounds__(256) void hblur(const float* __restrict__ x,
                                             const half_t* __restrict__ tf,
                                             half_t* __restrict__ y1t) {
  __shared__ half_t sm[16 * LSH];
  const int bc    = blockIdx.x;          // b*C + c
  const int b     = bc / C_;
  const int mtile = blockIdx.y;          // 0..23 -> padded rows i
  const int lane  = threadIdx.x & 31;
  const int wave  = threadIdx.x >> 5;    // 0..7
  const int lm    = lane & 15;
  const int hi    = lane >> 4;

  // 5 tap (B) fragments: aligned 32B vector loads from the pre-packed table
  v16h bfrag[5];
  const half_t* tb = tf + (size_t)b * 5 * TFS + lane * 16;
#pragma unroll
  for (int kk = 0; kk < 5; ++kk) bfrag[kk] = *(const v16h*)(tb + kk * TFS);

  // Stage 16 reflect-padded rows (f32 -> f16); zero the tail (taps are 0 there)
  const float* xc = x + (size_t)bc * (H_ * W_);
#pragma unroll
  for (int s = 0; s < 2; ++s) {
    int r = 2 * wave + s;
    int m = mtile * 16 + r; m = (m > HP_ - 1) ? (HP_ - 1) : m;
    const float* xrow = xc + (size_t)reflect_idx(m - R_, H_) * W_;
    for (int p = lane; p < LSH; p += 32) {
      half_t v = (half_t)0.0f;
      if (p < HP_) v = (half_t)xrow[reflect_idx(p - R_, W_)];
      sm[r * LSH + p] = v;
    }
  }
  __syncthreads();

  const size_t obase = (size_t)bc * 256 * IP_;
#pragma unroll
  for (int t = 0; t < 2; ++t) {
    const int w0 = (wave * 2 + t) * 16;
    v8f acc = {};
#pragma unroll
    for (int kk = 0; kk < 5; ++kk) {
      const half_t* pa = &sm[lm * LSH + w0 + 32 * kk + 8 * hi];
      v8h a0 = *(const v8h*)pa;            // K = hi*8 + 0..7
      v8h a1 = *(const v8h*)(pa + 16);     // K = 16 + hi*8 + 0..7
      v16h afrag = __builtin_shufflevector(a0, a1,
                     0,1,2,3,4,5,6,7,8,9,10,11,12,13,14,15);
      acc = __builtin_amdgcn_wmma_f32_16x16x32_f16(false, afrag, false, bfrag[kk],
                                                   (short)0, acc, false, false);
    }
    // D(M=v+8hi, N=lm): Y1t[w0+lm][mtile*16 + 8hi + v] -> contiguous 16B store
    v8h st;
#pragma unroll
    for (int v = 0; v < 8; ++v) st[v] = (half_t)acc[v];
    *(v8h*)(y1t + obase + (size_t)(w0 + lm) * IP_ + mtile * 16 + 8 * hi) = st;
  }
}

// ---------------------------------------------------------------------------
// Kernel 3: vertical pass. out[h][w] = sum_i Y1t[w][i] * k[i - h]
// A = Y1t rows staged to LDS via async global->LDS b128 copies (ASYNCcnt),
// B = same tap fragment table. D rows contiguous in w -> float4 stores.
// ---------------------------------------------------------------------------
__global__ __launch_bounds__(256) void vblur(const half_t* __restrict__ y1t,
                                             const half_t* __restrict__ tf,
                                             float* __restrict__ out) {
  __shared__ half_t sm[16 * LSV];
  const int bc    = blockIdx.x;
  const int b     = bc / C_;
  const int wtile = blockIdx.y;          // 0..15 -> w rows of Y1t
  const int lane  = threadIdx.x & 31;
  const int wave  = threadIdx.x >> 5;
  const int lm    = lane & 15;
  const int hi    = lane >> 4;

  v16h bfrag[5];
  const half_t* tb = tf + (size_t)b * 5 * TFS + lane * 16;
#pragma unroll
  for (int kk = 0; kk < 5; ++kk) bfrag[kk] = *(const v16h*)(tb + kk * TFS);

  // Stage 16 Y1t rows: 47 async b128 chunks per row (376 f16 = 47*16B),
  // zero-fill tail [376, LSV). Generic LDS pointer low 32 bits = LDS offset.
#pragma unroll
  for (int s = 0; s < 2; ++s) {
    int r = 2 * wave + s;
    const half_t* yrow = y1t + ((size_t)bc * 256 + wtile * 16 + r) * IP_;
    unsigned ldsrow = (unsigned)(uintptr_t)(sm + r * LSV);
    for (int q = lane; q < 47; q += 32) {
      unsigned ldsoff = ldsrow + (unsigned)(q * 16);
      const half_t* gsrc = yrow + q * 8;
      asm volatile("global_load_async_to_lds_b128 %0, %1, off"
                   :: "v"(ldsoff), "v"(gsrc) : "memory");
    }
    for (int i = HP_ + lane; i < LSV; i += 32) sm[r * LSV + i] = (half_t)0.0f;
  }
  asm volatile("s_wait_asynccnt 0x0" ::: "memory");
  __syncthreads();

  const size_t ob = (size_t)bc * H_ * W_;
#pragma unroll
  for (int t = 0; t < 2; ++t) {
    const int h0 = (wave * 2 + t) * 16;
    v8f acc = {};
#pragma unroll
    for (int kk = 0; kk < 5; ++kk) {
      const half_t* pa = &sm[lm * LSV + h0 + 32 * kk + 8 * hi];
      v8h a0 = *(const v8h*)pa;
      v8h a1 = *(const v8h*)(pa + 16);
      v16h afrag = __builtin_shufflevector(a0, a1,
                     0,1,2,3,4,5,6,7,8,9,10,11,12,13,14,15);
      acc = __builtin_amdgcn_wmma_f32_16x16x32_f16(false, afrag, false, bfrag[kk],
                                                   (short)0, acc, false, false);
    }
    // D(M=v+8hi -> w, N=lm -> h): out[h0+lm][wtile*16 + 8hi + v]
    float* po = out + ob + (size_t)(h0 + lm) * W_ + wtile * 16 + 8 * hi;
    float4 lo = make_float4(acc[0], acc[1], acc[2], acc[3]);
    float4 hi4 = make_float4(acc[4], acc[5], acc[6], acc[7]);
    *(float4*)po = lo;
    *(float4*)(po + 4) = hi4;
  }
}

// ---------------------------------------------------------------------------
// Launcher. Inputs: x (f32), fwd_steps (i32), blur_sigmas (f32). Output f32.
// Workspace: [0, 320KB) tap fragment table; [512KB, ~38MB) Y1t f16.
// ---------------------------------------------------------------------------
extern "C" void kernel_launch(void* const* d_in, const int* in_sizes, int n_in,
                              void* d_out, int out_size, void* d_ws, size_t ws_size,
                              hipStream_t stream) {
  const float* x   = (const float*)d_in[0];
  const int*   fwd = (const int*)d_in[1];
  const float* sig = (const float*)d_in[2];
  float*       out = (float*)d_out;

  half_t* tf  = (half_t*)d_ws;
  half_t* y1t = (half_t*)((char*)d_ws + (512 * 1024));

  build_taps<<<B_, 256, 0, stream>>>(fwd, sig, tf);
  hblur<<<dim3(B_ * C_, IP_ / 16), 256, 0, stream>>>(x, tf, y1t);
  vblur<<<dim3(B_ * C_, W_ / 16), 256, 0, stream>>>(y1t, tf, out);
}
